// ImplicitModel_88476326298065
// MI455X (gfx1250) — matrix operand order repaired and checked by reading, not compiled, not run on hardware
//
#include <hip/hip_runtime.h>
#include <math.h>

typedef __attribute__((ext_vector_type(16))) _Float16 v16h;
typedef __attribute__((ext_vector_type(8)))  float    v8f;
typedef __attribute__((ext_vector_type(4)))  unsigned int u32x4;

#define BATCH  4096
#define STATE  512
#define NONLIN 1024
#define ACTION 256
#define FITERS 30

union F16Frag { v16h v; u32x4 u[2]; };

// Force global (addrspace 1) 128-bit loads: avoids flat_load_b128 lowering,
// which would charge both LOADcnt and DScnt and consume LDS-path bandwidth.
// Uses a raw ext-vector type (not HIP's uint4 class) so the addrspace-
// qualified load is a builtin vector load on both host and device passes.
typedef const __attribute__((address_space(1))) u32x4* gu4p;
static __device__ __forceinline__ u32x4 gload(const _Float16* p) {
    return *(gu4p)p;
}

// ---------------------------------------------------------------------------
// elementwise f32 -> f16 (row-major A operands)
// ---------------------------------------------------------------------------
__global__ void f32_to_f16_kernel(const float* __restrict__ src,
                                  _Float16* __restrict__ dst, int n) {
    int i = blockIdx.x * blockDim.x + threadIdx.x;
    if (i < n) dst[i] = (_Float16)src[i];
}

// ---------------------------------------------------------------------------
// q0 = tanh(xc) as f16 (first fixed-point iteration with q = 0)
// ---------------------------------------------------------------------------
__global__ void tanh_f32_to_f16_kernel(const float* __restrict__ xc,
                                       _Float16* __restrict__ q, int n) {
    int i = blockIdx.x * blockDim.x + threadIdx.x;
    if (i < n) q[i] = (_Float16)tanhf(xc[i]);
}

// ---------------------------------------------------------------------------
// Pack a row-major f32 weight [K x N] into WMMA B-fragment order (f16).
// Per 32x16 (KxN) tile: 32 lanes x 8 dwords. Lane L (n = L&15) holds K block
// base (L<16 ? 0 : 16); dword j holds {B[base+2j][n], B[base+2j+1][n]}.
// Lane L's 16 f16 are contiguous -> the GEMM loads them as two 128-bit loads.
// ---------------------------------------------------------------------------
__global__ void pack_b_wmma_kernel(const float* __restrict__ src,
                                   _Float16* __restrict__ dst, int K, int N) {
    int t = blockIdx.x * blockDim.x + threadIdx.x;   // one dword (f16 pair)
    int total = (K * N) >> 1;
    if (t >= total) return;
    int tile   = t >> 8;            // 256 dwords per 32x16 tile
    int within = t & 255;
    int L = within >> 3;
    int j = within & 7;
    int tiles_n = N >> 4;
    int kt = tile / tiles_n;
    int nt = tile - kt * tiles_n;
    int kk = ((L >> 4) << 4) + (j << 1);     // lane half selects K block of 16
    int n  = (nt << 4) + (L & 15);
    size_t k0 = (size_t)(kt * 32 + kk);
    _Float16 lo = (_Float16)src[k0 * N + n];
    _Float16 hi = (_Float16)src[(k0 + 1) * N + n];
    dst[(size_t)2 * t]     = lo;
    dst[(size_t)2 * t + 1] = hi;
}

// ---------------------------------------------------------------------------
// WMMA GEMM: D[M x N] = op( bias + A[M x K] @ Bpacked[K x N] )
//   A:   f16 row-major
//   Bp:  f16 in packed B-fragment order (see pack_b_wmma_kernel)
//   bias: optional f32 [M x N]
//   op:  optional tanh
//   out: f32 (outF) or f16 row-major (outH)
// Block = 256 threads = 8 waves arranged 2(m) x 4(n); each wave computes a
// 64x64 output tile (4 x 4 fragments, 16 v_wmma per 32-deep K step); the
// block covers 128x256 of the output.
// ---------------------------------------------------------------------------
__launch_bounds__(256)
__global__ void gemm_wmma_kernel(const _Float16* __restrict__ A,
                                 const _Float16* __restrict__ Bp,
                                 const float* __restrict__ bias,
                                 float* __restrict__ outF,
                                 _Float16* __restrict__ outH,
                                 int M, int N, int K, int do_tanh) {
    const int lane    = threadIdx.x & 31;
    const int wid     = threadIdx.x >> 5;
    const int wave_m  = wid >> 2;                 // 0..1
    const int wave_n  = wid & 3;                  // 0..3
    const int row0    = blockIdx.y * 128 + wave_m * 64;
    const int col0    = blockIdx.x * 256 + wave_n * 64;
    const int lane_lo = lane & 15;
    const int half    = lane >> 4;
    const int tiles_n = N >> 4;
    const int ntile0  = col0 >> 4;

    const v8f vzero = {0.f, 0.f, 0.f, 0.f, 0.f, 0.f, 0.f, 0.f};
    v8f acc[4][4];
#pragma unroll
    for (int mi = 0; mi < 4; ++mi)
#pragma unroll
        for (int ni = 0; ni < 4; ++ni)
            acc[mi][ni] = vzero;

    // Per-lane loop-invariant A row base pointers (A layout: lane half ->
    // K base 0/16; VGPRs 0-7 cover K base+0..15).
    const _Float16* pA[4];
#pragma unroll
    for (int mi = 0; mi < 4; ++mi)
        pA[mi] = A + (size_t)(row0 + mi * 16 + lane_lo) * K + half * 16;

    // Packed-B: tile stride 512 f16; per k-step advance tiles_n tiles.
    const size_t bstep = (size_t)tiles_n * 512;
    const _Float16* pB = Bp + (size_t)ntile0 * 512 + lane * 16;

    const int ksteps = K >> 5;
    for (int kk = 0; kk < ksteps; ++kk) {
        F16Frag a[4];
#pragma unroll
        for (int mi = 0; mi < 4; ++mi) {
            const _Float16* p = pA[mi] + (size_t)kk * 32;
            a[mi].u[0] = gload(p);
            a[mi].u[1] = gload(p + 8);
        }
        F16Frag b[4];
#pragma unroll
        for (int ni = 0; ni < 4; ++ni) {
            const _Float16* p = pB + (size_t)kk * bstep + (size_t)ni * 512;
            b[ni].u[0] = gload(p);
            b[ni].u[1] = gload(p + 8);
            __builtin_prefetch((const void*)(p + bstep), 0, 1);
        }
#pragma unroll
        for (int mi = 0; mi < 4; ++mi)
#pragma unroll
            for (int ni = 0; ni < 4; ++ni)
                acc[mi][ni] = __builtin_amdgcn_wmma_f32_16x16x32_f16(
                    false, a[mi].v, false, b[ni].v,
                    (short)0, acc[mi][ni], false, false);
    }

    // Epilogue. C/D layout: lane L, VGPR r -> row = r + (L>=16 ? 8 : 0),
    // col = L & 15 within the 16x16 tile.
#pragma unroll
    for (int mi = 0; mi < 4; ++mi) {
#pragma unroll
        for (int ni = 0; ni < 4; ++ni) {
            const int col   = col0 + ni * 16 + lane_lo;
            const int rbase = row0 + mi * 16 + half * 8;
#pragma unroll
            for (int r = 0; r < 8; ++r) {
                size_t idx = (size_t)(rbase + r) * N + col;
                float v = acc[mi][ni][r];
                if (bias) v += bias[idx];
                if (do_tanh) v = tanhf(v);
                if (outH) outH[idx] = (_Float16)v;
                else      outF[idx] = v;
            }
        }
    }
}

// ---------------------------------------------------------------------------
extern "C" void kernel_launch(void* const* d_in, const int* in_sizes, int n_in,
                              void* d_out, int out_size, void* d_ws, size_t ws_size,
                              hipStream_t stream) {
    (void)in_sizes; (void)n_in; (void)out_size; (void)ws_size;
    const float* xs  = (const float*)d_in[0];
    const float* us  = (const float*)d_in[1];
    const float* A_T = (const float*)d_in[2];
    const float* B1T = (const float*)d_in[3];
    const float* B2T = (const float*)d_in[4];
    const float* C2T = (const float*)d_in[5];
    const float* D3T = (const float*)d_in[6];
    float* out = (float*)d_out;

    char* ws = (char*)d_ws;
    size_t off = 0;
    auto alloc = [&](size_t bytes) -> char* {
        char* p = ws + off;
        off += (bytes + 255) & ~(size_t)255;
        return p;
    };
    _Float16* xs_h = (_Float16*)alloc((size_t)BATCH * STATE * 2);
    _Float16* us_h = (_Float16*)alloc((size_t)BATCH * ACTION * 2);
    _Float16* Ap   = (_Float16*)alloc((size_t)STATE * STATE * 2);
    _Float16* B1p  = (_Float16*)alloc((size_t)NONLIN * STATE * 2);
    _Float16* B2p  = (_Float16*)alloc((size_t)ACTION * STATE * 2);
    _Float16* C2p  = (_Float16*)alloc((size_t)STATE * NONLIN * 2);
    _Float16* D3p  = (_Float16*)alloc((size_t)NONLIN * NONLIN * 2);
    float*    xc   = (float*)   alloc((size_t)BATCH * NONLIN * 4);
    float*    base = (float*)   alloc((size_t)BATCH * STATE * 4);
    _Float16* q0   = (_Float16*)alloc((size_t)BATCH * NONLIN * 2);
    _Float16* q1   = (_Float16*)alloc((size_t)BATCH * NONLIN * 2);

    int n;
    n = BATCH * STATE;
    f32_to_f16_kernel<<<(n + 255) / 256, 256, 0, stream>>>(xs, xs_h, n);
    n = BATCH * ACTION;
    f32_to_f16_kernel<<<(n + 255) / 256, 256, 0, stream>>>(us, us_h, n);

    n = (STATE * STATE) / 2;
    pack_b_wmma_kernel<<<(n + 255) / 256, 256, 0, stream>>>(A_T, Ap, STATE, STATE);
    n = (NONLIN * STATE) / 2;
    pack_b_wmma_kernel<<<(n + 255) / 256, 256, 0, stream>>>(B1T, B1p, NONLIN, STATE);
    n = (ACTION * STATE) / 2;
    pack_b_wmma_kernel<<<(n + 255) / 256, 256, 0, stream>>>(B2T, B2p, ACTION, STATE);
    n = (STATE * NONLIN) / 2;
    pack_b_wmma_kernel<<<(n + 255) / 256, 256, 0, stream>>>(C2T, C2p, STATE, NONLIN);
    n = (NONLIN * NONLIN) / 2;
    pack_b_wmma_kernel<<<(n + 255) / 256, 256, 0, stream>>>(D3T, D3p, NONLIN, NONLIN);

    const dim3 blk(256);
    // xc = xs @ C2_T   [4096 x 1024]
    gemm_wmma_kernel<<<dim3(NONLIN / 256, BATCH / 128), blk, 0, stream>>>(
        xs_h, C2p, nullptr, xc, nullptr, BATCH, NONLIN, STATE, 0);
    // q = tanh(xc)  (iteration 1, q_prev = 0)
    n = BATCH * NONLIN;
    tanh_f32_to_f16_kernel<<<(n + 255) / 256, 256, 0, stream>>>(xc, q0, n);
    // remaining 29 iterations: q = tanh(xc + q @ D3_T), ping-pong
    _Float16* qin = q0;
    _Float16* qout = q1;
    for (int it = 0; it < FITERS - 1; ++it) {
        gemm_wmma_kernel<<<dim3(NONLIN / 256, BATCH / 128), blk, 0, stream>>>(
            qin, D3p, xc, nullptr, qout, BATCH, NONLIN, NONLIN, 1);
        _Float16* t = qin; qin = qout; qout = t;
    }
    // base = xs @ A_T
    gemm_wmma_kernel<<<dim3(STATE / 256, BATCH / 128), blk, 0, stream>>>(
        xs_h, Ap, nullptr, base, nullptr, BATCH, STATE, STATE, 0);
    // base += us @ B2_T
    gemm_wmma_kernel<<<dim3(STATE / 256, BATCH / 128), blk, 0, stream>>>(
        us_h, B2p, base, base, nullptr, BATCH, STATE, ACTION, 0);
    // out = base + q* @ B1_T
    gemm_wmma_kernel<<<dim3(STATE / 256, BATCH / 128), blk, 0, stream>>>(
        qin, B1p, base, out, nullptr, BATCH, STATE, NONLIN, 0);
}
